// AddingProblemIndRNN_21294447853641
// MI455X (gfx1250) — matrix operand order repaired and checked by reading, not compiled, not run on hardware
//
#include <hip/hip_runtime.h>
#include <hip/hip_bf16.h>

// Problem constants (match reference)
#define BB 256
#define TT 2048
#define DD 2
#define HH 512
#define TC 64                    // time chunk
#define NCHUNK (TT / TC)         // 32
#define MCHUNK (BB * TC)         // 16384 rows per chunk GEMM

typedef __attribute__((ext_vector_type(16))) _Float16 v16h;
typedef __attribute__((ext_vector_type(8)))  _Float16 v8h;
typedef __attribute__((ext_vector_type(8)))  float    v8f;

// ---------------------------------------------------------------------------
// init: zero recurrent states, convert W2 (f32, [H,H] row-major g,h) -> fp16
// ---------------------------------------------------------------------------
__global__ void init_k(const float* __restrict__ W2, _Float16* __restrict__ W2h,
                       float* __restrict__ h1s, float* __restrict__ h2s) {
    int i = blockIdx.x * 256 + threadIdx.x;          // grid covers H*H = 262144
    W2h[i] = (_Float16)W2[i];
    if (i < BB * HH) { h1s[i] = 0.0f; h2s[i] = 0.0f; }
}

// ---------------------------------------------------------------------------
// scan1: fuse pre1 = x@W1^T + b1 (D=2) with layer-1 IndRNN scan over a chunk.
// One thread per (b,h) chain. Writes h1 chunk in fp16 (GEMM A operand).
// ---------------------------------------------------------------------------
__global__ void scan1_k(const float* __restrict__ x, const float* __restrict__ W1,
                        const float* __restrict__ u1, const float* __restrict__ b1,
                        float* __restrict__ h1s, _Float16* __restrict__ h1c, int chunk) {
    int i = blockIdx.x * 256 + threadIdx.x;          // B*H = 131072 threads
    int b = i >> 9;                                   // / H
    int h = i & (HH - 1);
    float w0 = W1[h * 2 + 0];
    float w1 = W1[h * 2 + 1];
    float u  = u1[h];
    float bias = b1[h];
    float st = h1s[i];
    const float* xp = x + ((size_t)b * TT + (size_t)chunk * TC) * DD;
    _Float16* op = h1c + (size_t)b * TC * HH + h;    // row (b*TC + t), col h
#pragma unroll 4
    for (int t = 0; t < TC; ++t) {
        float pre = fmaf(xp[2 * t], w0, fmaf(xp[2 * t + 1], w1, bias));
        st = pre + u * st;
        st = st > 0.0f ? st : 0.0f;                   // relu
        op[(size_t)t * HH] = (_Float16)st;
    }
    h1s[i] = st;
}

// ---------------------------------------------------------------------------
// WMMA GEMM: C[m][g] = sum_h A[m][h] * W2h[g][h]
//   A: fp16 row-major [MCHUNK x H]   (h1 chunk)
//   W2h: fp16 row-major [H x H]      (g major, h minor == B operand K-major)
//   C: f32 [MCHUNK x H]              (pre2, bias added later in scan2)
// Per-wave tile: 64x64 (4x4 WMMA frags) -> 16 wmma per 16 b128 loads per K=32
// Fragment layouts: 16-bit A 16x32 / B 32x16 use contiguous k (k = 16*(lane/16) + s) in both; f32 C 16x16 per ISA
// ---------------------------------------------------------------------------
__device__ inline v16h load_a_frag(const _Float16* __restrict__ A, int m0, int kbase, int lane) {
    int row = m0 + (lane & 15);
    int kb  = kbase + ((lane >> 4) << 4);            // lanes 16-31: +16
    const _Float16* p = A + (size_t)row * HH + kb;
    return *(const v16h*)p;                           // 16 contiguous K halves
}

__device__ inline v16h load_b_frag(const _Float16* __restrict__ W, int n0, int kbase, int lane) {
    int col = n0 + (lane & 15);                       // output g
    int kb  = kbase + ((lane >> 4) << 4);            // lanes 16-31: +16
    return *(const v16h*)(W + (size_t)col * HH + kb); // 16 contiguous K halves
}

__device__ inline void store_c(float* __restrict__ C, int m0, int n0, v8f c, int lane) {
    int row = m0 + ((lane >> 4) << 3);
    int col = n0 + (lane & 15);
#pragma unroll
    for (int r = 0; r < 8; ++r)
        C[(size_t)(row + r) * HH + col] = c[r];
}

__global__ __launch_bounds__(256) void gemm_k(const _Float16* __restrict__ A,
                                              const _Float16* __restrict__ W2h,
                                              float* __restrict__ Cout) {
    int lane = threadIdx.x & 31;
    int wave = threadIdx.x >> 5;                     // 0..7
    int m0 = blockIdx.x * 256 + (wave & 3) * 64;     // 4 waves along M
    int n0 = blockIdx.y * 128 + (wave >> 2) * 64;    // 2 waves along N

    v8f c[4][4];
#pragma unroll
    for (int i = 0; i < 4; ++i)
#pragma unroll
        for (int j = 0; j < 4; ++j) c[i][j] = {};

#pragma unroll 1
    for (int k = 0; k < HH; k += 32) {
        v16h a[4], b[4];
#pragma unroll
        for (int i = 0; i < 4; ++i) a[i] = load_a_frag(A, m0 + 16 * i, k, lane);
#pragma unroll
        for (int j = 0; j < 4; ++j) b[j] = load_b_frag(W2h, n0 + 16 * j, k, lane);
#pragma unroll
        for (int i = 0; i < 4; ++i)
#pragma unroll
            for (int j = 0; j < 4; ++j)
                c[i][j] = __builtin_amdgcn_wmma_f32_16x16x32_f16(
                    false, a[i], false, b[j], (short)0, c[i][j], false, false);
    }

#pragma unroll
    for (int i = 0; i < 4; ++i)
#pragma unroll
        for (int j = 0; j < 4; ++j)
            store_c(Cout, m0 + 16 * i, n0 + 16 * j, c[i][j], lane);
}

// ---------------------------------------------------------------------------
// scan2: layer-2 IndRNN scan over the chunk (keeps only running state)
// ---------------------------------------------------------------------------
__global__ void scan2_k(const float* __restrict__ pre2, const float* __restrict__ u2,
                        const float* __restrict__ b2, float* __restrict__ h2s) {
    int i = blockIdx.x * 256 + threadIdx.x;          // B*H threads
    int b = i >> 9;
    int g = i & (HH - 1);
    float u = u2[g];
    float bias = b2[g];
    float st = h2s[i];
    const float* pp = pre2 + (size_t)b * TC * HH + g;
#pragma unroll 4
    for (int t = 0; t < TC; ++t) {
        float p = pp[(size_t)t * HH] + bias;
        st = p + u * st;
        st = st > 0.0f ? st : 0.0f;
    }
    h2s[i] = st;
}

// ---------------------------------------------------------------------------
// final: out[b] = dot(hT[b,:], Wf) + bf
// ---------------------------------------------------------------------------
__global__ void final_k(const float* __restrict__ hT, const float* __restrict__ Wf,
                        const float* __restrict__ bf, float* __restrict__ out) {
    __shared__ float red[256];
    int b = blockIdx.x;
    int tid = threadIdx.x;
    float s = hT[(size_t)b * HH + tid] * Wf[tid]
            + hT[(size_t)b * HH + tid + 256] * Wf[tid + 256];
    red[tid] = s;
    __syncthreads();
    for (int off = 128; off > 0; off >>= 1) {
        if (tid < off) red[tid] += red[tid + off];
        __syncthreads();
    }
    if (tid == 0) out[b] = red[0] + bf[0];
}

// ---------------------------------------------------------------------------
// host launcher
// ---------------------------------------------------------------------------
extern "C" void kernel_launch(void* const* d_in, const int* in_sizes, int n_in,
                              void* d_out, int out_size, void* d_ws, size_t ws_size,
                              hipStream_t stream) {
    const float* x  = (const float*)d_in[0];
    const float* W1 = (const float*)d_in[1];
    const float* u1 = (const float*)d_in[2];
    const float* b1 = (const float*)d_in[3];
    const float* W2 = (const float*)d_in[4];
    const float* u2 = (const float*)d_in[5];
    const float* b2 = (const float*)d_in[6];
    const float* Wf = (const float*)d_in[7];
    const float* bf = (const float*)d_in[8];
    float* out = (float*)d_out;

    // workspace layout (256B aligned)
    char* ws = (char*)d_ws;
    size_t off = 0;
    auto take = [&](size_t bytes) { char* p = ws + off; off = (off + bytes + 255) & ~(size_t)255; return p; };
    float*    h1s   = (float*)   take((size_t)BB * HH * sizeof(float));        // 512 KB
    float*    h2s   = (float*)   take((size_t)BB * HH * sizeof(float));        // 512 KB
    _Float16* W2h   = (_Float16*)take((size_t)HH * HH * sizeof(_Float16));     // 512 KB
    _Float16* h1c   = (_Float16*)take((size_t)MCHUNK * HH * sizeof(_Float16)); // 16 MB
    float*    pre2c = (float*)   take((size_t)MCHUNK * HH * sizeof(float));    // 32 MB
    (void)ws_size; (void)in_sizes; (void)n_in; (void)out_size;

    // init states + fp16 weights
    init_k<<<dim3((HH * HH) / 256), dim3(256), 0, stream>>>(W2, W2h, h1s, h2s);

    dim3 scanGrid((BB * HH) / 256);
    dim3 gemmGrid(MCHUNK / 256, HH / 128);
    for (int c = 0; c < NCHUNK; ++c) {
        scan1_k<<<scanGrid, dim3(256), 0, stream>>>(x, W1, u1, b1, h1s, h1c, c);
        gemm_k <<<gemmGrid, dim3(256), 0, stream>>>(h1c, W2h, pre2c);
        scan2_k<<<scanGrid, dim3(256), 0, stream>>>(pre2c, u2, b2, h2s);
    }
    final_k<<<dim3(BB), dim3(256), 0, stream>>>(h2s, Wf, bf, out);
}
